// Model_26766236189372
// MI455X (gfx1250) — compile-verified, matrix operand order checked by reading, not run on hardware
//
#include <hip/hip_runtime.h>

typedef __attribute__((ext_vector_type(16))) _Float16 v16h;
typedef __attribute__((ext_vector_type(8)))  float    v8f;

#define N_NODES 10000
#define N_EDGES 40000
#define N_GRAPH 64
#define EDIM    4
#define HDIM    64
#define OUTD    4
#define K1      4096   // H*H

// ---------------------------------------------------------------------------
// Pack w2_1 (f32 [4096,4096], row-major k x n) into f16 WMMA-B tiles.
// Tile (kb in [0,128), nb in [0,256)) holds B[kb*32..+32, nb*16..+16].
// Within a tile: lane l (0..31) owns 16 contiguous halves; k pattern mirrors
// the documented 16-bit operand layout (lanes 0-15: K={0..7,16..23},
// lanes 16-31: K={8..15,24..31}).
// ---------------------------------------------------------------------------
__global__ __launch_bounds__(256) void pack_w2h(const float* __restrict__ w2,
                                                _Float16* __restrict__ out) {
  int idx = blockIdx.x * 256 + threadIdx.x;          // one half per thread
  if (idx >= 128 * 256 * 512) return;
  int r    = idx & 15;
  int l    = (idx >> 4) & 31;
  int tile = idx >> 9;
  int nb   = tile & 255;
  int kb   = tile >> 8;
  int n_in = l & 15;
  int k_in = (l < 16) ? ((r < 8) ? r : r + 8)
                      : ((r < 8) ? r + 8 : r + 16);
  out[idx] = (_Float16)w2[(kb * 32 + k_in) * K1 + nb * 16 + n_in];
}

// ---------------------------------------------------------------------------
// In-degree counts (shared by both NNConv layers).
// ---------------------------------------------------------------------------
__global__ void edge_count(const int* __restrict__ ei, float* __restrict__ cnt) {
  int e = blockIdx.x * 256 + threadIdx.x;
  if (e < N_EDGES) atomicAdd(&cnt[ei[N_EDGES + e]], 1.0f);
}

// ---------------------------------------------------------------------------
// Layer 0 edge kernel (in=1): h = relu(ea@w1+b1)@w2 + b2 ; msg = x[src]*h.
// 4 edges / block, 64 threads per edge (thread = output column).
// ---------------------------------------------------------------------------
__global__ __launch_bounds__(256) void nnconv0_edge(
    const int* __restrict__ ei, const float* __restrict__ x0,
    const float* __restrict__ ea, const float* __restrict__ w1,
    const float* __restrict__ b1, const float* __restrict__ w2,
    const float* __restrict__ b2, float* __restrict__ s0) {
  __shared__ float w2s[64 * 64];
  __shared__ float mid[4][64];
  __shared__ float xs[4];
  __shared__ int   dsts[4];
  int tid = threadIdx.x;
  for (int i = tid; i < 4096; i += 256) w2s[i] = w2[i];
  int eg = tid >> 6, o = tid & 63;
  int e  = blockIdx.x * 4 + eg;
  bool valid = (e < N_EDGES);
  float a0 = 0.f, a1 = 0.f, a2 = 0.f, a3 = 0.f;
  if (valid) {
    a0 = ea[e * 4 + 0]; a1 = ea[e * 4 + 1];
    a2 = ea[e * 4 + 2]; a3 = ea[e * 4 + 3];
    if (o == 0) { xs[eg] = x0[ei[e]]; dsts[eg] = ei[N_EDGES + e]; }
  }
  float m = b1[o] + a0 * w1[0 * 64 + o] + a1 * w1[1 * 64 + o] +
            a2 * w1[2 * 64 + o] + a3 * w1[3 * 64 + o];
  mid[eg][o] = fmaxf(m, 0.f);
  __syncthreads();
  if (valid) {
    float acc = b2[o];
#pragma unroll 8
    for (int j = 0; j < 64; ++j) acc += mid[eg][j] * w2s[j * 64 + o];
    atomicAdd(&s0[(size_t)dsts[eg] * 64 + o], xs[eg] * acc);
  }
}

// ---------------------------------------------------------------------------
// Layer 0 node update: x1 = relu(scatter_mean + x0*root + bias)
// ---------------------------------------------------------------------------
__global__ __launch_bounds__(256) void node0(
    const float* __restrict__ s0, const float* __restrict__ cnt,
    const float* __restrict__ x0, const float* __restrict__ root,
    const float* __restrict__ bias, float* __restrict__ x1) {
  int n = blockIdx.x * 4 + (threadIdx.x >> 6);
  int o = threadIdx.x & 63;
  if (n >= N_NODES) return;
  float c = fmaxf(cnt[n], 1.f);
  float v = s0[n * 64 + o] / c + x0[n] * root[o] + bias[o];
  x1[n * 64 + o] = fmaxf(v, 0.f);
}

// ---------------------------------------------------------------------------
// Layer 1 fused edge kernel — the hot loop.
// 16 edges / block, 8 waves. Per block:
//   1) Hm = relu(ea@w1_1+b1_1) -> [16,4096] f16 in LDS, WMMA-A layout (128 KB)
//   2) for each 128-col chunk: 8 waves x (128 k-steps of wmma_f32_16x16x32_f16)
//      -> C tile [16,128] f32 to LDS
//   3) contract C with x1[src] (+ b2_1 term) into per-thread msg accumulators
//   4) scatter-add messages into s1[dst]
// ---------------------------------------------------------------------------
__global__ __launch_bounds__(256) void nnconv1_edge(
    const int* __restrict__ ei, const float* __restrict__ ea,
    const float* __restrict__ x1, const float* __restrict__ w1,
    const float* __restrict__ b1, const _Float16* __restrict__ Bt,
    const float* __restrict__ b2, float* __restrict__ s1) {
  extern __shared__ v16h dynsmem[];            // forces 32B alignment
  char* smem = (char*)dynsmem;
  _Float16* Alds = (_Float16*)smem;            // 128*512 halves  = 131072 B
  float*    Clds = (float*)(smem + 131072);    // 16*128 f32      =   8192 B
  float*    x1s  = (float*)(smem + 131072 + 8192);        // 16*64 = 4096 B
  float*    eas  = (float*)(smem + 131072 + 8192 + 4096); // 16*4  =  256 B
  __shared__ int sdst[16];

  int tid = threadIdx.x;
  int e0  = blockIdx.x * 16;

  if (tid < 64) eas[tid] = ea[e0 * 4 + tid];
  if (tid < 16) sdst[tid] = ei[N_EDGES + e0 + tid];
  {
    int i = tid & 63;
#pragma unroll
    for (int j = 0; j < 4; ++j) {
      int ee  = (tid >> 6) + j * 4;
      int src = ei[e0 + ee];
      x1s[ee * 64 + i] = x1[(size_t)src * 64 + i];
    }
  }
  __syncthreads();

  // Hm tile directly in WMMA-A layout: Alds[kb*512 + l*16 + r]
  for (int idx = tid; idx < 128 * 512; idx += 256) {
    int r   = idx & 15;
    int l   = (idx >> 4) & 31;
    int kb  = idx >> 9;
    int e   = l & 15;
    int kin = (l < 16) ? ((r < 8) ? r : r + 8) : ((r < 8) ? r + 8 : r + 16);
    int k   = kb * 32 + kin;
    float v = b1[k];
#pragma unroll
    for (int d = 0; d < 4; ++d) v += eas[e * 4 + d] * w1[d * K1 + k];
    Alds[idx] = (_Float16)fmaxf(v, 0.f);
  }
  __syncthreads();

  int lane = tid & 31, wave = tid >> 5;
  int o    = tid & 63;
  float msg[4] = {0.f, 0.f, 0.f, 0.f};

  for (int chunk = 0; chunk < 32; ++chunk) {
    // ---- GEMM: wave computes 16x16 tile at columns chunk*128 + wave*16 ----
    v8f acc = {};
    const _Float16* bp = Bt + (size_t)(chunk * 8 + wave) * 512 + lane * 16;
#pragma unroll 4
    for (int kb = 0; kb < 128; ++kb) {
      v16h a = *(const v16h*)(Alds + kb * 512 + lane * 16);
      v16h b = *(const v16h*)bp;
      bp += 256 * 512;                                  // next kb tile row
      __builtin_prefetch(bp + 4 * 256 * 512, 0, 1);     // global_prefetch_b8
      acc = __builtin_amdgcn_wmma_f32_16x16x32_f16(
          false, a, false, b, (short)0, acc, false, false);
    }
    // ---- C tile -> LDS (doc layout: VGPR r, lanes<16 -> M=r, else M=r+8) ----
    int col   = wave * 16 + (lane & 15);
    int mbase = (lane < 16) ? 0 : 8;
#pragma unroll
    for (int r = 0; r < 8; ++r) Clds[(mbase + r) * 128 + col] = acc[r];
    __syncthreads();
    // ---- contract with x1[src]: msg[e,o] += x1[src,i]*(C + b2_1) ----
    int i0   = chunk * 2;
    float ba = b2[i0 * 64 + o];
    float bb = b2[(i0 + 1) * 64 + o];
#pragma unroll
    for (int j = 0; j < 4; ++j) {
      int e    = (tid >> 6) + j * 4;
      float xa = x1s[e * 64 + i0];
      float xb = x1s[e * 64 + i0 + 1];
      msg[j] += xa * (Clds[e * 128 + o] + ba) +
                xb * (Clds[e * 128 + 64 + o] + bb);
    }
    __syncthreads();
  }
#pragma unroll
  for (int j = 0; j < 4; ++j) {
    int e = (tid >> 6) + j * 4;
    atomicAdd(&s1[(size_t)sdst[e] * 64 + o], msg[j]);
  }
}

// ---------------------------------------------------------------------------
// Layer 1 node update + graph mean-pool accumulation.
// ---------------------------------------------------------------------------
__global__ __launch_bounds__(256) void node1(
    const float* __restrict__ s1, const float* __restrict__ cnt,
    const float* __restrict__ x1, const float* __restrict__ root,
    const float* __restrict__ bias, const int* __restrict__ batch,
    float* __restrict__ gsum, float* __restrict__ gcnt) {
  __shared__ float roots[64 * 64];
  __shared__ float xls[4][64];
  int tid = threadIdx.x;
  for (int i = tid; i < 4096; i += 256) roots[i] = root[i];
  int ng = tid >> 6, o = tid & 63;
  int n  = blockIdx.x * 4 + ng;
  if (n < N_NODES) xls[ng][o] = x1[(size_t)n * 64 + o];
  __syncthreads();
  if (n >= N_NODES) return;
  float acc = bias[o];
#pragma unroll 8
  for (int i = 0; i < 64; ++i) acc += xls[ng][i] * roots[i * 64 + o];
  float c = fmaxf(cnt[n], 1.f);
  float v = fmaxf(s1[(size_t)n * 64 + o] / c + acc, 0.f);
  int b = batch[n];
  atomicAdd(&gsum[b * 64 + o], v);
  if (o == 0) atomicAdd(&gcnt[b], 1.f);
}

// ---------------------------------------------------------------------------
// Classifier head: 64 graphs, one thread each.
// ---------------------------------------------------------------------------
__global__ __launch_bounds__(64) void classifier(
    const float* __restrict__ gsum, const float* __restrict__ gcnt,
    const float* __restrict__ wc1, const float* __restrict__ bc1,
    const float* __restrict__ wc2, const float* __restrict__ bc2,
    float* __restrict__ out) {
  int g = threadIdx.x;
  float pooled[64], h[64];
  float c = fmaxf(gcnt[g], 1.f);
  for (int i = 0; i < 64; ++i) pooled[i] = gsum[g * 64 + i] / c;
  for (int k = 0; k < 64; ++k) {
    float a = bc1[k];
    for (int i = 0; i < 64; ++i) a += pooled[i] * wc1[i * 64 + k];
    h[k] = fmaxf(a, 0.f);
  }
  for (int o = 0; o < OUTD; ++o) {
    float a = bc2[o];
    for (int k = 0; k < 64; ++k) a += h[k] * wc2[k * OUTD + o];
    out[g * OUTD + o] = a;
  }
}

// ---------------------------------------------------------------------------
extern "C" void kernel_launch(void* const* d_in, const int* in_sizes, int n_in,
                              void* d_out, int out_size, void* d_ws,
                              size_t ws_size, hipStream_t stream) {
  const float* x0    = (const float*)d_in[0];
  const int*   ei    = (const int*)d_in[1];
  const float* ea    = (const float*)d_in[2];
  const int*   batch = (const int*)d_in[3];
  const float* w1_0  = (const float*)d_in[4];
  const float* b1_0  = (const float*)d_in[5];
  const float* w2_0  = (const float*)d_in[6];
  const float* b2_0  = (const float*)d_in[7];
  const float* root0 = (const float*)d_in[8];
  const float* bias0 = (const float*)d_in[9];
  const float* w1_1  = (const float*)d_in[10];
  const float* b1_1  = (const float*)d_in[11];
  const float* w2_1  = (const float*)d_in[12];
  const float* b2_1  = (const float*)d_in[13];
  const float* root1 = (const float*)d_in[14];
  const float* bias1 = (const float*)d_in[15];
  const float* wc1   = (const float*)d_in[16];
  const float* bc1   = (const float*)d_in[17];
  const float* wc2   = (const float*)d_in[18];
  const float* bc2   = (const float*)d_in[19];
  float* out = (float*)d_out;

  // Workspace carve-up (256B aligned)
  char* ws = (char*)d_ws;
  size_t off = 0;
  auto carve = [&](size_t bytes) {
    void* p = ws + off;
    off = (off + bytes + 255) & ~(size_t)255;
    return p;
  };
  _Float16* w2h = (_Float16*)carve((size_t)K1 * K1 * sizeof(_Float16)); // 33.5MB
  float* s0   = (float*)carve((size_t)N_NODES * HDIM * 4);
  float* s1   = (float*)carve((size_t)N_NODES * HDIM * 4);
  float* x1   = (float*)carve((size_t)N_NODES * HDIM * 4);
  float* cnt  = (float*)carve((size_t)N_NODES * 4);
  float* gsum = (float*)carve((size_t)N_GRAPH * HDIM * 4);
  float* gcnt = (float*)carve((size_t)N_GRAPH * 4);

  hipMemsetAsync(s0, 0, (size_t)N_NODES * HDIM * 4, stream);
  hipMemsetAsync(s1, 0, (size_t)N_NODES * HDIM * 4, stream);
  hipMemsetAsync(cnt, 0, (size_t)N_NODES * 4, stream);
  hipMemsetAsync(gsum, 0, (size_t)N_GRAPH * HDIM * 4, stream);
  hipMemsetAsync(gcnt, 0, (size_t)N_GRAPH * 4, stream);

  // Pack w2_1 -> f16 WMMA-B tiles (B fits entirely in 192MB L2 afterwards).
  pack_w2h<<<(128 * 256 * 512) / 256, 256, 0, stream>>>(w2_1, w2h);

  edge_count<<<(N_EDGES + 255) / 256, 256, 0, stream>>>(ei, cnt);

  // Layer 0
  nnconv0_edge<<<(N_EDGES + 3) / 4, 256, 0, stream>>>(ei, x0, ea, w1_0, b1_0,
                                                      w2_0, b2_0, s0);
  node0<<<(N_NODES + 3) / 4, 256, 0, stream>>>(s0, cnt, x0, root0, bias0, x1);

  // Layer 1 — fused WMMA GEMM + message contraction + scatter
  size_t shbytes = 131072 + 8192 + 4096 + 256;  // ~140.5 KB of CDNA5 LDS
  nnconv1_edge<<<N_EDGES / 16, 256, shbytes, stream>>>(ei, ea, x1, w1_1, b1_1,
                                                       w2h, b2_1, s1);
  node1<<<(N_NODES + 3) / 4, 256, 0, stream>>>(s1, cnt, x1, root1, bias1,
                                               batch, gsum, gcnt);

  // Head
  classifier<<<1, 64, 0, stream>>>(gsum, gcnt, wc1, bc1, wc2, bc2, out);
}
